// MultiheadAttention3d_47682726921138
// MI455X (gfx1250) — matrix operand order repaired and checked by reading, not compiled, not run on hardware
//
#include <hip/hip_runtime.h>
#include <math.h>

typedef _Float16 h16;
typedef __attribute__((ext_vector_type(16))) _Float16 v16h;
typedef __attribute__((ext_vector_type(8)))  _Float16 v8h;
typedef __attribute__((ext_vector_type(8)))  float    v8f;

#define NTOK 4096
#define HD   64
#define NHD  8
#define HID  512
#define QSCALE 0.04419417382415922f   // 1/sqrt(512)  (faithful: sqrt(hidden))

__device__ __forceinline__ unsigned lane_id() { return threadIdx.x & 31u; }

__device__ __forceinline__ v8f wmma16(v16h a, v16h b, v8f c) {
  // D(16x16 f32) = A(16x32 f16) x B(32x16 f16) + C
  return __builtin_amdgcn_wmma_f32_16x16x32_f16(false, a, false, b, (short)0, c,
                                                false, false);
}

// ---- A fragment (16x32 f16), source row-major, leading dim ld ----
__device__ __forceinline__ v16h load_a_h(const h16* __restrict__ src, int ld) {
  unsigned l = lane_id();
  unsigned m = l & 15u;
  unsigned koff = (l & 16u) ? 16u : 0u;
  const h16* row = src + (size_t)m * ld + koff;
  v8h lo = *(const v8h*)(row);
  v8h hi = *(const v8h*)(row + 8);
  v16h a;
#pragma unroll
  for (int i = 0; i < 8; ++i) { a[i] = lo[i]; a[8 + i] = hi[i]; }
  return a;
}

// A fragment from f32 source with convert
__device__ __forceinline__ v16h load_a_f(const float* __restrict__ src, int ld) {
  unsigned l = lane_id();
  unsigned m = l & 15u;
  unsigned koff = (l & 16u) ? 16u : 0u;
  const float* row = src + (size_t)m * ld + koff;
  v16h a;
#pragma unroll
  for (int i = 0; i < 8; ++i) { a[i] = (h16)row[i]; a[8 + i] = (h16)row[8 + i]; }
  return a;
}

// ---- B fragment (32x16): B[k,n] = src[n*ld + k] (src rows = N dim) ----
__device__ __forceinline__ v16h load_bt_h(const h16* __restrict__ src, int ld) {
  unsigned l = lane_id();
  unsigned n = l & 15u;
  unsigned koff = (l & 16u) ? 16u : 0u;
  const h16* row = src + (size_t)n * ld + koff;
  v8h lo = *(const v8h*)(row);
  v8h hi = *(const v8h*)(row + 8);
  v16h b;
#pragma unroll
  for (int i = 0; i < 8; ++i) { b[i] = lo[i]; b[8 + i] = hi[i]; }
  return b;
}

__device__ __forceinline__ v16h load_bt_f(const float* __restrict__ src, int ld) {
  unsigned l = lane_id();
  unsigned n = l & 15u;
  unsigned koff = (l & 16u) ? 16u : 0u;
  const float* row = src + (size_t)n * ld + koff;
  v16h b;
#pragma unroll
  for (int i = 0; i < 16; ++i) b[i] = (h16)row[i];
  return b;
}

// =====================================================================
// Kernel 1: QKV GEMM (M=4096,K=64,N=1536) + bias + layout-scramble scatter
// =====================================================================
__global__ void qkv_gemm_scatter(const float* __restrict__ x,
                                 const float* __restrict__ w_qkv,
                                 const float* __restrict__ b_qkv,
                                 h16* __restrict__ qbuf,
                                 h16* __restrict__ kbuf,
                                 h16* __restrict__ vtb) {
  unsigned wid  = threadIdx.x >> 5;
  unsigned tile = blockIdx.x * 8u + wid;       // 24576 tiles total
  unsigned tm = tile / 96u, tn = tile % 96u;   // 256 M-tiles x 96 N-tiles
  unsigned m0 = tm * 16u, n0 = tn * 16u;

  v8f acc = {};
  v16h a0 = load_a_f(x + (size_t)m0 * 64, 64);
  v16h a1 = load_a_f(x + (size_t)m0 * 64 + 32, 64);
  v16h b0 = load_bt_f(w_qkv + (size_t)n0 * 64, 64);       // B[k,n]=W[o=n][k]
  v16h b1 = load_bt_f(w_qkv + (size_t)n0 * 64 + 32, 64);
  acc = wmma16(a0, b0, acc);
  acc = wmma16(a1, b1, acc);

  unsigned l = lane_id();
  unsigned n = l & 15u;
  unsigned mh = (l & 16u) ? 8u : 0u;
  unsigned o = n0 + n;
  float bias = b_qkv[o];
  unsigned s   = o >> 9;          // 0:q 1:k 2:v
  unsigned nh  = (o >> 6) & 7u;
  unsigned hdp = o & 63u;
#pragma unroll
  for (int r = 0; r < 8; ++r) {
    unsigned mlin = m0 + (unsigned)r + mh;          // = c*64 + d*8 + h
    unsigned dp = mlin >> 9, hp = (mlin >> 6) & 7u, wp = mlin & 63u;
    unsigned mtok = (dp * 64u + hdp) * 8u + hp;     // scrambled token index
    float v = acc[r] + bias;                         // feature index = wp
    if (s == 0)
      qbuf[((size_t)nh * NTOK + mtok) * HD + wp] = (h16)(v * QSCALE);
    else if (s == 1)
      kbuf[((size_t)nh * NTOK + mtok) * HD + wp] = (h16)v;
    else
      vtb[((size_t)nh * HD + wp) * NTOK + mtok] = (h16)v;   // V transposed
  }
}

// =====================================================================
// Kernel 2: flash attention.  1 WG (8 waves) = 256 query rows of one head,
// 32 rows per wave.  K/V 32-key tiles are cooperatively staged in LDS with
// double buffering so all 8 waves share one K/V stream (8x less L2 traffic
// than private streaming).  Online softmax; scramble-scatter epilogue.
// =====================================================================
__global__ void __launch_bounds__(256)
flash_attn(const h16* __restrict__ qbuf,
           const h16* __restrict__ kbuf,
           const h16* __restrict__ vtb,
           h16* __restrict__ obuf) {
  __shared__ __align__(16) h16 Kl[2][32][64];     // [buf][key][hd]     8 KB
  __shared__ __align__(16) h16 Vl[2][64][32];     // [buf][hd][key]     8 KB
  __shared__ __align__(16) h16 Pbuf[8][16][32];   // per-wave P stage   8 KB

  unsigned tid = threadIdx.x;
  unsigned wid = tid >> 5;
  unsigned bh  = blockIdx.x >> 4;                  // head (b=1), 16 WGs/head
  unsigned m0  = (blockIdx.x & 15u) * 256u + wid * 32u;

  const h16* kb = kbuf + (size_t)bh * NTOK * HD;
  const h16* vb = vtb  + (size_t)bh * HD * NTOK;

  // Q fragments: two 16-row tiles per wave, K-dim 64 -> 2 frags each
  v16h qa[2][2];
#pragma unroll
  for (int u = 0; u < 2; ++u) {
    const h16* qp = qbuf + ((size_t)bh * NTOK + m0 + u * 16) * HD;
    qa[u][0] = load_a_h(qp, HD);
    qa[u][1] = load_a_h(qp + 32, HD);
  }

  unsigned l = lane_id();
  unsigned nlane = l & 15u;
  unsigned mh = (l & 16u) ? 8u : 0u;

  float rmax[2][8], rsum[2][8];
  v8f zero = {};
  v8f oacc[2][4];
#pragma unroll
  for (int u = 0; u < 2; ++u)
#pragma unroll
    for (int r = 0; r < 8; ++r) { rmax[u][r] = -INFINITY; rsum[u][r] = 0.f; }
#pragma unroll
  for (int u = 0; u < 2; ++u)
#pragma unroll
    for (int c = 0; c < 4; ++c) oacc[u][c] = zero;

  // cooperative staging addresses: 256 threads x 16B = one 4 KB tile each
  unsigned krow = tid >> 3, kcol = (tid & 7u) * 8u;   // K: 32 x 64
  unsigned vrow = tid >> 2, vcol = (tid & 3u) * 8u;   // V: 64 x 32

  // prologue: stage tile 0
  {
    v8h kr = *(const v8h*)(kb + (size_t)krow * HD + kcol);
    v8h vr = *(const v8h*)(vb + (size_t)vrow * NTOK + vcol);
    *(v8h*)&Kl[0][krow][kcol] = kr;
    *(v8h*)&Vl[0][vrow][vcol] = vr;
  }
  __syncthreads();

  for (int kt = 0; kt < 128; ++kt) {
    int kv0 = kt * 32;
    int cur = kt & 1, nxt = cur ^ 1;

    // issue global loads for tile kt+1 early (latency hiding)
    v8h kr = {}, vr = {};
    bool have = (kt + 1) < 128;
    if (have) {
      kr = *(const v8h*)(kb + (size_t)(kv0 + 32 + krow) * HD + kcol);
      vr = *(const v8h*)(vb + (size_t)vrow * NTOK + kv0 + 32 + vcol);
      if (kt + 2 < 128) {                 // pull tile kt+2 toward L2/L0
        __builtin_prefetch(kb + (size_t)(kv0 + 64 + krow) * HD + kcol, 0, 1);
        __builtin_prefetch(vb + (size_t)vrow * NTOK + kv0 + 64 + vcol, 0, 1);
      }
    }

    // K/V fragments from LDS, shared by both Q row-tiles of this wave
    v16h kf0 = load_bt_h(&Kl[cur][0][0],  64);
    v16h kf1 = load_bt_h(&Kl[cur][0][32], 64);
    v16h kf2 = load_bt_h(&Kl[cur][16][0], 64);
    v16h kf3 = load_bt_h(&Kl[cur][16][32],64);
    v16h vf[4];
#pragma unroll
    for (int c = 0; c < 4; ++c) vf[c] = load_bt_h(&Vl[cur][c * 16][0], 32);

#pragma unroll
    for (int u = 0; u < 2; ++u) {
      // S = Q * K^T (16 x 32 keys, f32 acc); Q pre-scaled by 1/sqrt(512)
      v8f slo = wmma16(qa[u][0], kf0, zero);
      slo     = wmma16(qa[u][1], kf1, slo);
      v8f shi = wmma16(qa[u][0], kf2, zero);
      shi     = wmma16(qa[u][1], kf3, shi);

      // online softmax: row stats across the 16 lanes of each half-wave
      float nmax[8], corr[8];
#pragma unroll
      for (int r = 0; r < 8; ++r) {
        float t = fmaxf(slo[r], shi[r]);
        t = fmaxf(t, __shfl_xor(t, 1));
        t = fmaxf(t, __shfl_xor(t, 2));
        t = fmaxf(t, __shfl_xor(t, 4));
        t = fmaxf(t, __shfl_xor(t, 8));
        nmax[r] = fmaxf(rmax[u][r], t);
        corr[r] = __expf(rmax[u][r] - nmax[r]);
        rmax[u][r] = nmax[r];
      }
#pragma unroll
      for (int c = 0; c < 4; ++c)
#pragma unroll
        for (int r = 0; r < 8; ++r) oacc[u][c][r] *= corr[r];

#pragma unroll
      for (int r = 0; r < 8; ++r) {
        float plo = __expf(slo[r] - nmax[r]);
        float phi = __expf(shi[r] - nmax[r]);
        float ts = plo + phi;
        ts += __shfl_xor(ts, 1);
        ts += __shfl_xor(ts, 2);
        ts += __shfl_xor(ts, 4);
        ts += __shfl_xor(ts, 8);
        rsum[u][r] = rsum[u][r] * corr[r] + ts;
        unsigned row = (unsigned)r + mh;        // D-layout -> LDS transpose
        Pbuf[wid][row][nlane]      = (h16)plo;
        Pbuf[wid][row][16 + nlane] = (h16)phi;
      }
      asm volatile("s_wait_dscnt 0" ::: "memory");   // LDS RAW + reorder fence
      v16h pa = load_a_h(&Pbuf[wid][0][0], 32);      // re-read in A layout

#pragma unroll
      for (int c = 0; c < 4; ++c) oacc[u][c] = wmma16(pa, vf[c], oacc[u][c]);
    }

    // stage tile kt+1 into the other buffer; reads of it happened before the
    // previous end-of-iteration barrier, so this is safe.
    if (have) {
      *(v8h*)&Kl[nxt][krow][kcol] = kr;
      *(v8h*)&Vl[nxt][vrow][vcol] = vr;
    }
    __syncthreads();
  }

  // epilogue: out(bh, f, mtok) -> scrambled (n2, hidden) for projection
#pragma unroll
  for (int u = 0; u < 2; ++u)
#pragma unroll
    for (int c = 0; c < 4; ++c) {
      unsigned f = (unsigned)c * 16u + nlane;
#pragma unroll
      for (int r = 0; r < 8; ++r) {
        unsigned mtok = m0 + (unsigned)u * 16u + (unsigned)r + mh;
        float v = oacc[u][c][r] / rsum[u][r];
        unsigned flat = f * (unsigned)NTOK + mtok;
        unsigned hdpp = flat & 63u;
        unsigned wpp  = (flat >> 6) & 63u;
        unsigned hpp  = (flat >> 12) & 7u;
        unsigned dpp  = flat >> 15;
        unsigned n2  = ((bh * 8u + dpp) * 8u + hpp) * 8u + (wpp >> 3);
        unsigned hid = (wpp & 7u) * 64u + hdpp;
        obuf[(size_t)n2 * HID + hid] = (h16)v;
      }
    }
}

// =====================================================================
// Kernel 3: projection GEMM (M=4096,K=512,N=64) + bias -> f32 d_out
// =====================================================================
__global__ void proj_gemm(const h16* __restrict__ obuf,
                          const float* __restrict__ w_proj,
                          const float* __restrict__ b_proj,
                          float* __restrict__ out) {
  unsigned wid  = threadIdx.x >> 5;
  unsigned tile = blockIdx.x * 8u + wid;     // 1024 tiles
  unsigned tm = tile >> 2, tn = tile & 3u;   // 256 x 4
  unsigned m0 = tm * 16u, n0 = tn * 16u;

  v8f acc = {};
#pragma unroll 4
  for (int kk = 0; kk < 16; ++kk) {
    int k0 = kk * 32;
    v16h a = load_a_h(obuf + (size_t)m0 * HID + k0, HID);
    v16h b = load_bt_f(w_proj + (size_t)n0 * HID + k0, HID);
    acc = wmma16(a, b, acc);
  }
  unsigned l = lane_id();
  unsigned n = l & 15u;
  unsigned mh = (l & 16u) ? 8u : 0u;
  float bias = b_proj[n0 + n];
#pragma unroll
  for (int r = 0; r < 8; ++r) {
    unsigned m = m0 + (unsigned)r + mh;      // final reshape is layout-identity
    out[(size_t)m * 64 + n0 + n] = acc[r] + bias;
  }
}

extern "C" void kernel_launch(void* const* d_in, const int* in_sizes, int n_in,
                              void* d_out, int out_size, void* d_ws, size_t ws_size,
                              hipStream_t stream) {
  (void)in_sizes; (void)n_in; (void)out_size; (void)ws_size;
  const float* x      = (const float*)d_in[0];
  const float* w_qkv  = (const float*)d_in[1];
  const float* b_qkv  = (const float*)d_in[2];
  const float* w_proj = (const float*)d_in[3];
  const float* b_proj = (const float*)d_in[4];
  float* out = (float*)d_out;

  // workspace carve (f16): q, k, vT : 8*4096*64 each ; attn-out: 4096*512
  h16* qbuf = (h16*)d_ws;
  h16* kbuf = qbuf + (size_t)NHD * NTOK * HD;
  h16* vtb  = kbuf + (size_t)NHD * NTOK * HD;
  h16* obuf = vtb  + (size_t)NHD * NTOK * HD;   // total 16 MB

  qkv_gemm_scatter<<<3072, 256, 0, stream>>>(x, w_qkv, b_qkv, qbuf, kbuf, vtb);
  flash_attn<<<128, 256, 0, stream>>>(qbuf, kbuf, vtb, obuf);
  proj_gemm<<<128, 256, 0, stream>>>(obuf, w_proj, b_proj, out);
}
